// GAT_5239860101747
// MI455X (gfx1250) — compile-verified, matrix operand order checked by reading, not run on hardware
//
#include <hip/hip_runtime.h>

// ---------------- problem constants (from reference) ----------------
#define NN  100000      // nodes
#define NF  256         // input features
#define HH  8           // heads (layer 1)
#define CC  16          // hidden per head
#define HC  128         // HH*CC
#define NC  16          // classes
#define EE  1600000     // raw edges
#define ET  1700000     // edges + self loops
#define NEG 0.2f

typedef _Float16 h8   __attribute__((ext_vector_type(8)));
typedef _Float16 v16h __attribute__((ext_vector_type(16)));
typedef float    v8f  __attribute__((ext_vector_type(8)));

// ---------------- helpers ----------------
__device__ __forceinline__ void edge_sd(const int* __restrict__ ei, int e, int& s, int& d) {
    if (e < EE) { s = ei[e]; d = ei[EE + e]; }
    else        { s = e - EE; d = s; }
}

// loop-free float atomic max via ordered int/uint atomics (init must be -inf)
__device__ __forceinline__ void atomicMaxF(float* addr, float val) {
    if (val >= 0.0f) atomicMax((int*)addr, __float_as_int(val));
    else             atomicMin((unsigned int*)addr, __float_as_uint(val));
}

__device__ __forceinline__ float lrelu(float a) { return a > 0.0f ? a : NEG * a; }

// ---------------- weight / input conversion to f16 ----------------
__global__ __launch_bounds__(256) void gat_cvtw(const float* __restrict__ W1,
                                                const float* __restrict__ W2,
                                                _Float16* __restrict__ w1h,
                                                _Float16* __restrict__ w2h) {
    int i = blockIdx.x * 256 + threadIdx.x;
    if (i < HC * NF)              w1h[i] = (_Float16)W1[i];
    else if (i < HC * NF + NC*HC) w2h[i - HC * NF] = (_Float16)W2[i - HC * NF];
}

__global__ __launch_bounds__(256) void gat_cvtx(const float* __restrict__ x,
                                                _Float16* __restrict__ xh) {
    int i = blockIdx.x * 256 + threadIdx.x;
    if (i < NN * NF) xh[i] = (_Float16)x[i];
}

// ---------------- GEMM1: h1[N,128] = xh[N,256] @ W1h[128,256]^T ----------------
// W1h staged to LDS once per block (64 KB) with 16B-chunk XOR swizzle:
// chunk c of row r stored at c^(r&15)  -> B-frag reads hit 16 distinct bank groups.
// One wave computes a 16x128 row strip. CDNA5 16-bit fragment layout:
// lane l (<16): row l, K = kb..kb+7 / kb+16..kb+23 ; lane l+16: K = kb+8.. / kb+24..
__global__ __launch_bounds__(256)
void gat_gemm1(const _Float16* __restrict__ xh, const _Float16* __restrict__ w1h,
               float* __restrict__ h1) {
    __shared__ _Float16 lw[HC * NF];                 // exactly 64 KB
    const int tid = threadIdx.x;
    // stage: 128 rows x 32 chunks of 8 halves (16 B)
    for (int idx = tid; idx < HC * 32; idx += 256) {
        int r = idx >> 5, ch = idx & 31;
        *(h8*)(&lw[r * NF + ((ch ^ (r & 15)) << 3)]) =
            *(const h8*)(w1h + (size_t)r * NF + (ch << 3));
    }
    __syncthreads();

    const int lane = tid & 31;
    const int rt   = blockIdx.x * 8 + (tid >> 5);
    if (rt >= NN / 16) return;                       // wave-uniform after barrier
    const int rsub = lane & 15;
    const int ksel = (lane >> 4) << 3;               // 0 or 8 (halves)
    const _Float16* xrow = xh + (size_t)(rt * 16 + rsub) * NF;

    v8f acc[8] = {};
    #pragma unroll
    for (int ks = 0; ks < 8; ++ks) {
        const int kb = ks * 32 + ksel;               // in halves
        const int c0 = kb >> 3;                      // 16B-chunk index
        union { v16h v; h8 p[2]; } A;
        A.p[0] = *(const h8*)(xrow + kb);
        A.p[1] = *(const h8*)(xrow + kb + 16);
        #pragma unroll
        for (int ct = 0; ct < 8; ++ct) {
            const int n = ct * 16 + rsub;
            union { v16h v; h8 p[2]; } B;
            B.p[0] = *(const h8*)(&lw[n * NF + (((c0    ) ^ (n & 15)) << 3)]);
            B.p[1] = *(const h8*)(&lw[n * NF + (((c0 + 2) ^ (n & 15)) << 3)]);
            acc[ct] = __builtin_amdgcn_wmma_f32_16x16x32_f16(
                false, A.v, false, B.v, (short)0, acc[ct], false, false);
        }
    }
    // D layout: VGPR r, lane l -> row = r + 8*(l>>4), col = l&15
    const int rhi = (lane >> 4) << 3;
    #pragma unroll
    for (int ct = 0; ct < 8; ++ct) {
        #pragma unroll
        for (int r = 0; r < 8; ++r)
            h1[(size_t)(rt * 16 + rhi + r) * HC + ct * 16 + rsub] = acc[ct][r];
    }
}

// ---------------- attention scalars + accumulator init (layer 1) ----------------
__global__ __launch_bounds__(256)
void gat_att1(const float* __restrict__ h1, const float* __restrict__ as,
              const float* __restrict__ ad, float* __restrict__ a_src,
              float* __restrict__ a_dst, float* __restrict__ amax,
              float* __restrict__ den, float* __restrict__ agg) {
    int i = blockIdx.x * 256 + threadIdx.x;
    if (i >= NN * HH) return;
    int n = i >> 3, h = i & 7;
    const float* hp = h1 + (size_t)n * HC + h * CC;
    float s = 0.0f, d = 0.0f;
    #pragma unroll
    for (int c = 0; c < CC; ++c) { s += hp[c] * as[h * CC + c]; d += hp[c] * ad[h * CC + c]; }
    a_src[i] = s; a_dst[i] = d;
    amax[i] = -__builtin_inff(); den[i] = 0.0f;
    float* ap = agg + (size_t)n * HC + h * CC;
    #pragma unroll
    for (int c = 0; c < CC; ++c) ap[c] = 0.0f;
}

// ---------------- segment softmax passes (layer 1) ----------------
__global__ __launch_bounds__(256)
void gat_emax1(const int* __restrict__ ei, const float* __restrict__ a_src,
               const float* __restrict__ a_dst, float* __restrict__ amax) {
    int i = blockIdx.x * 256 + threadIdx.x;
    if (i >= ET * HH) return;
    int e = i >> 3, h = i & 7, s, d;
    edge_sd(ei, e, s, d);
    atomicMaxF(&amax[d * HH + h], lrelu(a_src[s * HH + h] + a_dst[d * HH + h]));
}

__global__ __launch_bounds__(256)
void gat_eden1(const int* __restrict__ ei, const float* __restrict__ a_src,
               const float* __restrict__ a_dst, const float* __restrict__ amax,
               float* __restrict__ den) {
    int i = blockIdx.x * 256 + threadIdx.x;
    if (i >= ET * HH) return;
    int e = i >> 3, h = i & 7, s, d;
    edge_sd(ei, e, s, d);
    float al = lrelu(a_src[s * HH + h] + a_dst[d * HH + h]);
    atomicAdd(&den[d * HH + h], expf(al - amax[d * HH + h]));
}

// materialize per-edge attention weights once (kills 16x exp recompute in agg pass)
__global__ __launch_bounds__(256)
void gat_attw1(const int* __restrict__ ei, const float* __restrict__ a_src,
               const float* __restrict__ a_dst, const float* __restrict__ amax,
               const float* __restrict__ den, float* __restrict__ attn) {
    int i = blockIdx.x * 256 + threadIdx.x;
    if (i >= ET * HH) return;
    int e = i >> 3, h = i & 7, s, d;
    edge_sd(ei, e, s, d);
    float al = lrelu(a_src[s * HH + h] + a_dst[d * HH + h]);
    attn[i] = expf(al - amax[d * HH + h]) / (den[d * HH + h] + 1e-16f);
}

// one wave = one edge (32 lanes x 4 channels): edge_sd loads are wave-uniform
__global__ __launch_bounds__(256)
void gat_eagg1(const int* __restrict__ ei, const float* __restrict__ attn,
               const float* __restrict__ h1, float* __restrict__ agg) {
    int i = blockIdx.x * 256 + threadIdx.x;   // < ET*32
    int e = i >> 5, q = i & 31;               // q = channel quad
    if (e >= ET) return;
    int c = q << 2;
    int s, d; edge_sd(ei, e, s, d);
    float a = attn[e * HH + (q >> 2)];
    const float4 hv = *(const float4*)(h1 + (size_t)s * HC + c);
    float* ap = agg + (size_t)d * HC + c;
    atomicAdd(ap + 0, hv.x * a);
    atomicAdd(ap + 1, hv.y * a);
    atomicAdd(ap + 2, hv.z * a);
    atomicAdd(ap + 3, hv.w * a);
}

__global__ __launch_bounds__(256)
void gat_biaselu(float* __restrict__ agg, const float* __restrict__ b1) {
    int i = blockIdx.x * 256 + threadIdx.x;
    if (i >= NN * HC) return;
    float v = agg[i] + b1[i & 127];
    agg[i] = v > 0.0f ? v : (expf(v) - 1.0f);     // ELU(alpha=1)
}

// ---------------- GEMM2: h2[N,16] = out1[N,128] @ W2h[16,128]^T ----------------
__global__ __launch_bounds__(256)
void gat_gemm2(const float* __restrict__ a1, const _Float16* __restrict__ w2h,
               float* __restrict__ h2) {
    const int lane = threadIdx.x & 31;
    const int rt   = blockIdx.x * 8 + (threadIdx.x >> 5);
    if (rt >= NN / 16) return;
    const int rsub = lane & 15;
    const int ksel = (lane >> 4) << 3;
    const float*    arow = a1 + (size_t)(rt * 16 + rsub) * HC;
    const _Float16* wrow = w2h + (size_t)rsub * HC;

    v8f acc = {};
    #pragma unroll
    for (int ks = 0; ks < 4; ++ks) {
        const int kb = ks * 32 + ksel;
        union { v16h v; _Float16 e[16]; } A;
        #pragma unroll
        for (int t = 0; t < 8; ++t) {
            A.e[t]     = (_Float16)arow[kb + t];
            A.e[8 + t] = (_Float16)arow[kb + 16 + t];
        }
        union { v16h v; h8 p[2]; } B;
        B.p[0] = *(const h8*)(wrow + kb);
        B.p[1] = *(const h8*)(wrow + kb + 16);
        acc = __builtin_amdgcn_wmma_f32_16x16x32_f16(
            false, A.v, false, B.v, (short)0, acc, false, false);
    }
    const int rhi = (lane >> 4) << 3;
    #pragma unroll
    for (int r = 0; r < 8; ++r)
        h2[(size_t)(rt * 16 + rhi + r) * NC + rsub] = acc[r];
}

// ---------------- layer 2 attention (heads = 1) ----------------
__global__ __launch_bounds__(256)
void gat_att2(const float* __restrict__ h2, const float* __restrict__ as,
              const float* __restrict__ ad, float* __restrict__ a_src,
              float* __restrict__ a_dst, float* __restrict__ amax,
              float* __restrict__ den, float* __restrict__ agg) {
    int n = blockIdx.x * 256 + threadIdx.x;
    if (n >= NN) return;
    const float* hp = h2 + (size_t)n * NC;
    float s = 0.0f, d = 0.0f;
    #pragma unroll
    for (int c = 0; c < NC; ++c) { s += hp[c] * as[c]; d += hp[c] * ad[c]; }
    a_src[n] = s; a_dst[n] = d;
    amax[n] = -__builtin_inff(); den[n] = 0.0f;
    float* ap = agg + (size_t)n * NC;
    #pragma unroll
    for (int c = 0; c < NC; ++c) ap[c] = 0.0f;
}

__global__ __launch_bounds__(256)
void gat_emax2(const int* __restrict__ ei, const float* __restrict__ a_src,
               const float* __restrict__ a_dst, float* __restrict__ amax) {
    int e = blockIdx.x * 256 + threadIdx.x;
    if (e >= ET) return;
    int s, d; edge_sd(ei, e, s, d);
    atomicMaxF(&amax[d], lrelu(a_src[s] + a_dst[d]));
}

__global__ __launch_bounds__(256)
void gat_eden2(const int* __restrict__ ei, const float* __restrict__ a_src,
               const float* __restrict__ a_dst, const float* __restrict__ amax,
               float* __restrict__ den) {
    int e = blockIdx.x * 256 + threadIdx.x;
    if (e >= ET) return;
    int s, d; edge_sd(ei, e, s, d);
    float al = lrelu(a_src[s] + a_dst[d]);
    atomicAdd(&den[d], expf(al - amax[d]));
}

__global__ __launch_bounds__(256)
void gat_attw2(const int* __restrict__ ei, const float* __restrict__ a_src,
               const float* __restrict__ a_dst, const float* __restrict__ amax,
               const float* __restrict__ den, float* __restrict__ attn) {
    int e = blockIdx.x * 256 + threadIdx.x;
    if (e >= ET) return;
    int s, d; edge_sd(ei, e, s, d);
    float al = lrelu(a_src[s] + a_dst[d]);
    attn[e] = expf(al - amax[d]) / (den[d] + 1e-16f);
}

__global__ __launch_bounds__(256)
void gat_eagg2(const int* __restrict__ ei, const float* __restrict__ attn,
               const float* __restrict__ h2, float* __restrict__ agg) {
    int i = blockIdx.x * 256 + threadIdx.x;   // < ET*4
    int e = i >> 2, q = i & 3;
    if (e >= ET) return;
    int c = q << 2;
    int s, d; edge_sd(ei, e, s, d);
    float a = attn[e];
    const float4 hv = *(const float4*)(h2 + (size_t)s * NC + c);
    float* ap = agg + (size_t)d * NC + c;
    atomicAdd(ap + 0, hv.x * a);
    atomicAdd(ap + 1, hv.y * a);
    atomicAdd(ap + 2, hv.z * a);
    atomicAdd(ap + 3, hv.w * a);
}

// ---------------- bias + log_softmax ----------------
__global__ __launch_bounds__(256)
void gat_lsm(const float* __restrict__ agg, const float* __restrict__ b2,
             float* __restrict__ out) {
    int n = blockIdx.x * 256 + threadIdx.x;
    if (n >= NN) return;
    float v[NC], m = -__builtin_inff();
    #pragma unroll
    for (int c = 0; c < NC; ++c) { v[c] = agg[(size_t)n * NC + c] + b2[c]; m = fmaxf(m, v[c]); }
    float sum = 0.0f;
    #pragma unroll
    for (int c = 0; c < NC; ++c) sum += expf(v[c] - m);
    float ls = logf(sum);
    #pragma unroll
    for (int c = 0; c < NC; ++c) out[(size_t)n * NC + c] = v[c] - m - ls;
}

// ---------------- host orchestration ----------------
extern "C" void kernel_launch(void* const* d_in, const int* in_sizes, int n_in,
                              void* d_out, int out_size, void* d_ws, size_t ws_size,
                              hipStream_t stream) {
    (void)in_sizes; (void)n_in; (void)out_size; (void)ws_size;
    const float* x   = (const float*)d_in[0];
    const int*   ei  = (const int*)d_in[1];
    const float* W1  = (const float*)d_in[2];
    const float* as1 = (const float*)d_in[3];
    const float* ad1 = (const float*)d_in[4];
    const float* b1  = (const float*)d_in[5];
    const float* W2  = (const float*)d_in[6];
    const float* as2 = (const float*)d_in[7];
    const float* ad2 = (const float*)d_in[8];
    const float* b2  = (const float*)d_in[9];
    float* out = (float*)d_out;

    char* wsb = (char*)d_ws;
    size_t off = 0;
    auto alloc = [&](size_t bytes) -> void* {
        void* p = wsb + off;
        off += (bytes + 255) & ~(size_t)255;
        return p;
    };
    _Float16* w1h  = (_Float16*)alloc((size_t)HC * NF * 2);
    _Float16* w2h  = (_Float16*)alloc((size_t)NC * HC * 2);
    _Float16* xh   = (_Float16*)alloc((size_t)NN * NF * 2);   // 51.2 MB, dead after gemm1
    float* h1      = (float*)alloc((size_t)NN * HC * 4);
    float* asrc1   = (float*)alloc((size_t)NN * HH * 4);
    float* adst1   = (float*)alloc((size_t)NN * HH * 4);
    float* amax1   = (float*)alloc((size_t)NN * HH * 4);
    float* den1    = (float*)alloc((size_t)NN * HH * 4);
    float* attn1   = (float*)alloc((size_t)ET * HH * 4);      // 54.4 MB
    float* h2      = (float*)alloc((size_t)NN * NC * 4);
    float* asrc2   = (float*)alloc((size_t)NN * 4);
    float* adst2   = (float*)alloc((size_t)NN * 4);
    float* amax2   = (float*)alloc((size_t)NN * 4);
    float* den2    = (float*)alloc((size_t)NN * 4);
    float* attn2   = (float*)alloc((size_t)ET * 4);
    float* agg2    = (float*)alloc((size_t)NN * NC * 4);
    float* agg1    = (float*)xh;   // alias: N*HC*4 == N*NF*2 bytes; xh consumed by gemm1 first

    const dim3 B(256);
    // layer 1
    gat_cvtw <<<(HC * NF + NC * HC + 255) / 256, B, 0, stream>>>(W1, W2, w1h, w2h);
    gat_cvtx <<<(NN * NF) / 256, B, 0, stream>>>(x, xh);
    gat_gemm1<<<(NN / 16 + 7) / 8, B, 0, stream>>>(xh, w1h, h1);
    gat_att1 <<<(NN * HH) / 256, B, 0, stream>>>(h1, as1, ad1, asrc1, adst1, amax1, den1, agg1);
    gat_emax1<<<(ET * HH) / 256, B, 0, stream>>>(ei, asrc1, adst1, amax1);
    gat_eden1<<<(ET * HH) / 256, B, 0, stream>>>(ei, asrc1, adst1, amax1, den1);
    gat_attw1<<<(ET * HH) / 256, B, 0, stream>>>(ei, asrc1, adst1, amax1, den1, attn1);
    gat_eagg1<<<(ET * 32) / 256, B, 0, stream>>>(ei, attn1, h1, agg1);
    gat_biaselu<<<(NN * HC) / 256, B, 0, stream>>>(agg1, b1);
    // layer 2
    gat_gemm2<<<(NN / 16 + 7) / 8, B, 0, stream>>>(agg1, w2h, h2);
    gat_att2 <<<(NN + 255) / 256, B, 0, stream>>>(h2, as2, ad2, asrc2, adst2, amax2, den2, agg2);
    gat_emax2<<<(ET + 255) / 256, B, 0, stream>>>(ei, asrc2, adst2, amax2);
    gat_eden2<<<(ET + 255) / 256, B, 0, stream>>>(ei, asrc2, adst2, amax2, den2);
    gat_attw2<<<(ET + 255) / 256, B, 0, stream>>>(ei, asrc2, adst2, amax2, den2, attn2);
    gat_eagg2<<<(ET * 4 + 255) / 256, B, 0, stream>>>(ei, attn2, h2, agg2);
    gat_lsm  <<<(NN + 255) / 256, B, 0, stream>>>(agg2, b2, out);
}
